// MixtureOfExpertsEP_49443663512012
// MI455X (gfx1250) — matrix operand order repaired and checked by reading, not compiled, not run on hardware
//
#include <hip/hip_runtime.h>
#include <cstdint>
#include <cstddef>

// Problem constants (from reference)
#define D_MODEL 2048
#define D_FF    8192
#define NEXP    8
#define TOPK    2
#define SEQTOK  8192                 // S = B*T
#define CAP     2560                 // ceil(1.25*S*K/E)
#define SK      (SEQTOK * TOPK)      // 16384
#define SENT    SK

typedef __bf16 bf16_t;
typedef __bf16 v2bf  __attribute__((ext_vector_type(2)));
typedef __bf16 v4bf  __attribute__((ext_vector_type(4)));
typedef __bf16 v16bf __attribute__((ext_vector_type(16)));
typedef float  v8f   __attribute__((ext_vector_type(8)));

// ---------- helpers (native cvt: RNE, lowers to v_cvt[_pk]_bf16_f32) ----------
__device__ __forceinline__ uint32_t pack_bf16x2(float lo, float hi) {
  v2bf v;
  v[0] = (bf16_t)lo;
  v[1] = (bf16_t)hi;
  union { v2bf v; uint32_t u; } c; c.v = v;
  return c.u;
}

// ---------- 0) zero output ----------
__global__ void __launch_bounds__(256) zero_kernel(float* __restrict__ p, int n) {
  int i = blockIdx.x * 256 + threadIdx.x;
  if (i < n) p[i] = 0.0f;
}

// ---------- 1) gate: logits -> softmax -> top-2 ----------
__global__ void __launch_bounds__(256) gate_topk_kernel(
    const float* __restrict__ x, const float* __restrict__ Wg,
    float* __restrict__ gate_vals, int* __restrict__ expert_idx) {
  const int lane  = threadIdx.x & 31;
  const int wave  = threadIdx.x >> 5;
  const int token = blockIdx.x * 8 + wave;
  const float* xrow = x + (size_t)token * D_MODEL;

  float acc[NEXP];
#pragma unroll
  for (int e = 0; e < NEXP; ++e) acc[e] = 0.0f;

  for (int k = lane; k < D_MODEL; k += 32) {
    float xv = xrow[k];
    const float* wrow = Wg + (size_t)k * NEXP;
#pragma unroll
    for (int e = 0; e < NEXP; ++e) acc[e] += xv * wrow[e];
  }
#pragma unroll
  for (int off = 16; off > 0; off >>= 1) {
#pragma unroll
    for (int e = 0; e < NEXP; ++e) acc[e] += __shfl_xor(acc[e], off, 32);
  }

  if (lane == 0) {
    float m = acc[0];
#pragma unroll
    for (int e = 1; e < NEXP; ++e) m = fmaxf(m, acc[e]);
    float p[NEXP]; float sum = 0.0f;
#pragma unroll
    for (int e = 0; e < NEXP; ++e) { p[e] = __expf(acc[e] - m); sum += p[e]; }
    float inv = 1.0f / sum;
    int i0 = 0; float v0 = p[0];
#pragma unroll
    for (int e = 1; e < NEXP; ++e) if (p[e] > v0) { v0 = p[e]; i0 = e; }
    int i1 = -1; float v1 = -1.0f;
#pragma unroll
    for (int e = 0; e < NEXP; ++e) if (e != i0 && p[e] > v1) { v1 = p[e]; i1 = e; }
    gate_vals[token * TOPK + 0]  = v0 * inv;
    gate_vals[token * TOPK + 1]  = v1 * inv;
    expert_idx[token * TOPK + 0] = i0;
    expert_idx[token * TOPK + 1] = i1;
  }
}

// ---------- 2) route: order-preserving capacity assignment ----------
__global__ void __launch_bounds__(256) route_kernel(
    const int* __restrict__ expert_idx, int* __restrict__ slotTable) {
  const int lane = threadIdx.x & 31;
  const int e    = threadIdx.x >> 5;

  for (int c = lane; c < CAP; c += 32) slotTable[e * CAP + c] = SENT;

  int count = 0;
  for (int base = 0; base < SK; base += 32) {
    int i = base + lane;
    bool m = (expert_idx[i] == e);
    unsigned mask = __builtin_amdgcn_ballot_w32(m);
    if (m) {
      int pos = count + __popc(mask & ((1u << lane) - 1u));
      if (pos < CAP) {
        int s = i >> 1, kk = i & 1;            // flat i = s*K + k
        slotTable[e * CAP + pos] = kk * SEQTOK + s;
      }
    }
    count += __popc(mask);
  }
}

// ---------- 3) gather -> bf16 dispatched[E*CAP, D_MODEL] ----------
__global__ void __launch_bounds__(256) gather_kernel(
    const float* __restrict__ x, const int* __restrict__ slotTable,
    bf16_t* __restrict__ dispatched) {
  const int row = blockIdx.x;                   // e*CAP + c
  int slot = slotTable[row];
  int s = slot & (SEQTOK - 1);                  // SENT % S == 0 (matches reference)
  const float4* src = (const float4*)(x + (size_t)s * D_MODEL);
  uint32_t* dst = (uint32_t*)(dispatched + (size_t)row * D_MODEL);
  for (int j = threadIdx.x; j < D_MODEL / 4; j += 256) {
    float4 v = src[j];
    dst[2 * j + 0] = pack_bf16x2(v.x, v.y);
    dst[2 * j + 1] = pack_bf16x2(v.z, v.w);
  }
}

// ---------- 4/5) WMMA GEMM: 128x128 tile, 8 waves, K-step 32, double-buffered ----------
// EPI==0: D = relu(A@B) stored bf16 to Hout.  EPI==1: fused combine epilogue.
template <int EPI>
__global__ void __launch_bounds__(256) gemm_kernel(
    const bf16_t* __restrict__ A,      // [E, CAP, Kdim] row-major (bf16)
    const float*  __restrict__ Bw,     // [E, Kdim, N]  row-major (f32)
    bf16_t* __restrict__ Hout,         // EPI==0 dest: [E, CAP, N] bf16
    float* __restrict__ out,           // EPI==1 dest: [S, D_MODEL] f32
    const int* __restrict__ slotTable,
    const float* __restrict__ gate_vals,
    int Kdim, int N) {
  // A tile: stored in natural K order so a wave A-fragment is one contiguous 32B LDS read.
  //   positions: k -> k (lane half h reads k 16h..16h+15)
  // B tile: stored [khalf][n][ke] so a wave B-fragment is one contiguous 32B LDS read.
  __shared__ __align__(64) bf16_t As[2][128][32];       // 16 KB
  __shared__ __align__(64) bf16_t Bs[2][2][128][16];    // 16 KB

  const int lane   = threadIdx.x & 31;
  const int waveId = threadIdx.x >> 5;
  const int wm = waveId >> 2;            // 0..1  (64-row wave tile)
  const int wn = waveId & 3;             // 0..3  (32-col wave tile)
  const int lane16 = lane & 15;
  const int hi = (lane < 16) ? 0 : 1;

  const int e  = blockIdx.z;
  const int m0 = blockIdx.y * 128;
  const int n0 = blockIdx.x * 128;

  const bf16_t* Ae = A  + (size_t)e * CAP  * Kdim;
  const float*  Be = Bw + (size_t)e * Kdim * N;

  const v8f vzero = {0.f, 0.f, 0.f, 0.f, 0.f, 0.f, 0.f, 0.f};
  v8f acc[4][2];
#pragma unroll
  for (int mi = 0; mi < 4; ++mi)
#pragma unroll
    for (int ni = 0; ni < 2; ++ni) acc[mi][ni] = vzero;

  // staging thread mappings
  const int tA_row  = threadIdx.x >> 1;          // 0..127
  const int tA_half = threadIdx.x & 1;           // which 16-wide K chunk
  const int nq = threadIdx.x & 31;               // B: 4-col group (0..31)
  const int kq = threadIdx.x >> 5;               // B: 4-row group (0..7)

  // register staging (global -> regs -> LDS), enables load/WMMA overlap
  uint4 ra0, ra1;
  float rb[4][4];

  auto load_tile = [&](int k0) {
    const uint4* ag = (const uint4*)(Ae + (size_t)(m0 + tA_row) * Kdim + k0 + tA_half * 16);
    ra0 = ag[0];                                 // 8 bf16: chunk k +0..7
    ra1 = ag[1];                                 // 8 bf16: chunk k +8..15
    const float* bbase = Be + (size_t)(k0 + 4 * kq) * N + n0 + 4 * nq;
#pragma unroll
    for (int r = 0; r < 4; ++r) {
      float4 v = *(const float4*)(bbase + (size_t)r * N);
      rb[r][0] = v.x; rb[r][1] = v.y; rb[r][2] = v.z; rb[r][3] = v.w;
    }
  };
  auto store_tile = [&](int p) {
    bf16_t* dstrow = &As[p][tA_row][0];
    if (tA_half == 0) { *(uint4*)(dstrow + 0) = ra0; *(uint4*)(dstrow + 8) = ra1; }
    else              { *(uint4*)(dstrow + 16) = ra0; *(uint4*)(dstrow + 24) = ra1; }
#pragma unroll
    for (int c = 0; c < 4; ++c) {
      v4bf pk;                                   // packed cvt: v_cvt_pk_bf16_f32 x2
      pk[0] = (bf16_t)rb[0][c];
      pk[1] = (bf16_t)rb[1][c];
      pk[2] = (bf16_t)rb[2][c];
      pk[3] = (bf16_t)rb[3][c];
      *(v4bf*)&Bs[p][kq >> 2][4 * nq + c][4 * (kq & 3)] = pk;
    }
  };

  load_tile(0);
  int p = 0;
  for (int k0 = 0; k0 < Kdim; k0 += 32) {
    store_tile(p);
    __syncthreads();                            // one barrier per K-step
    if (k0 + 32 < Kdim) {
      load_tile(k0 + 32);                       // overlap next-tile HBM reads with WMMA
      if (k0 + 128 < Kdim)                      // stream weights ahead (global_prefetch_b8)
        __builtin_prefetch(Be + (size_t)(k0 + 128 + 4 * kq) * N + n0 + 4 * nq, 0, 1);
    }

    // B fragments: N = lane%16, K = 16*hi + elem  -> contiguous 32B read
    v16bf bfrag[2];
#pragma unroll
    for (int ni = 0; ni < 2; ++ni)
      bfrag[ni] = *(const v16bf*)&Bs[p][hi][wn * 32 + ni * 16 + lane16][0];

    // A fragments + WMMA
#pragma unroll
    for (int mi = 0; mi < 4; ++mi) {
      const v16bf a = *(const v16bf*)&As[p][wm * 64 + mi * 16 + lane16][hi * 16];
#pragma unroll
      for (int ni = 0; ni < 2; ++ni) {
        acc[mi][ni] = __builtin_amdgcn_wmma_f32_16x16x32_bf16(
            false, a, false, bfrag[ni], (short)0, acc[mi][ni], false, false);
      }
    }
    p ^= 1;
  }

  // epilogue: C/D layout M = v + 8*hi, N = lane%16
#pragma unroll
  for (int mi = 0; mi < 4; ++mi) {
#pragma unroll
    for (int v = 0; v < 8; ++v) {
      const int row = m0 + wm * 64 + mi * 16 + v + hi * 8;
      if (EPI == 0) {
#pragma unroll
        for (int ni = 0; ni < 2; ++ni) {
          const int col = n0 + wn * 32 + ni * 16 + lane16;
          float val = fmaxf(acc[mi][ni][v], 0.0f);       // ReLU
          Hout[((size_t)e * CAP + row) * N + col] = (bf16_t)val;
        }
      } else {
        const int slot = slotTable[e * CAP + row];
        if (slot < SK) {
          const int s  = slot & (SEQTOK - 1);
          const int kk = slot >> 13;                     // slot / SEQTOK
          const float w = gate_vals[s * TOPK + kk];
#pragma unroll
          for (int ni = 0; ni < 2; ++ni) {
            const int col = n0 + wn * 32 + ni * 16 + lane16;
            atomicAdd(&out[(size_t)s * D_MODEL + col], w * acc[mi][ni][v]);
          }
        }
      }
    }
  }
}

// ---------- launch ----------
extern "C" void kernel_launch(void* const* d_in, const int* in_sizes, int n_in,
                              void* d_out, int out_size, void* d_ws, size_t ws_size,
                              hipStream_t stream) {
  (void)in_sizes; (void)n_in; (void)ws_size;
  const float* x  = (const float*)d_in[0];
  const float* Wg = (const float*)d_in[1];
  const float* w1 = (const float*)d_in[2];
  const float* w2 = (const float*)d_in[3];
  float* out = (float*)d_out;

  char* ws = (char*)d_ws;
  size_t off = 0;
  auto carve = [&](size_t bytes) -> void* {
    void* p = (void*)(ws + off);
    off += (bytes + 255) & ~(size_t)255;
    return p;
  };
  float*  gate_vals  = (float*)carve(sizeof(float) * SK);
  int*    expert_idx = (int*)carve(sizeof(int) * SK);
  int*    slotTable  = (int*)carve(sizeof(int) * NEXP * CAP);
  bf16_t* dispatched = (bf16_t*)carve(sizeof(uint16_t) * (size_t)NEXP * CAP * D_MODEL);
  bf16_t* h          = (bf16_t*)carve(sizeof(uint16_t) * (size_t)NEXP * CAP * D_FF);

  zero_kernel<<<(out_size + 255) / 256, 256, 0, stream>>>(out, out_size);
  gate_topk_kernel<<<SEQTOK / 8, 256, 0, stream>>>(x, Wg, gate_vals, expert_idx);
  route_kernel<<<1, 256, 0, stream>>>(expert_idx, slotTable);
  gather_kernel<<<NEXP * CAP, 256, 0, stream>>>(x, slotTable, dispatched);

  dim3 g1(D_FF / 128, CAP / 128, NEXP);      // 64 x 20 x 8
  gemm_kernel<0><<<g1, 256, 0, stream>>>(dispatched, w1, h, nullptr,
                                         nullptr, nullptr, D_MODEL, D_FF);
  dim3 g2(D_MODEL / 128, CAP / 128, NEXP);   // 16 x 20 x 8
  gemm_kernel<1><<<g2, 256, 0, stream>>>(h, w2, nullptr, out,
                                         slotTable, gate_vals, D_FF, D_MODEL);
}
